// BlockDiagonal_88570815578416
// MI455X (gfx1250) — matrix operand order repaired and checked by reading, not compiled, not run on hardware
//
#include <hip/hip_runtime.h>

typedef __attribute__((ext_vector_type(16))) _Float16 v16h;
typedef __attribute__((ext_vector_type(8)))  _Float16 v8h;
typedef __attribute__((ext_vector_type(4)))  _Float16 v4h;
typedef __attribute__((ext_vector_type(8)))  float    v8f;
typedef __attribute__((ext_vector_type(4)))  unsigned v4u;
typedef __attribute__((ext_vector_type(8)))  int      v8i;
typedef __attribute__((ext_vector_type(4)))  int      v4i;

#define LAYER    4096
#define BS       256        // block size (16 diagonal blocks of 256x256)
#define NBLK     16
#define M_TILE   128        // output rows per workgroup
#define N_TILE   256        // full block width per workgroup -> x read exactly once
#define K_CHUNK  64         // K staged in LDS per pass (4 passes over BS)
#define LSTRIDE  72         // halves/row: 144B = 16B-aligned rows, bank-conflict-free

#if __has_builtin(__builtin_amdgcn_tensor_load_to_lds)
#define USE_TDM 1
#else
#define USE_TDM 0
#endif

// ---------------- kernel 1: wsb[b][n][k] = (f16) tanh(B_b[k][n]) ----------------
__global__ __launch_bounds__(256)
void prep_bt_f16(const float* __restrict__ blocks, _Float16* __restrict__ wsb)
{
    const int b  = blockIdx.y;                 // diagonal block
    const int k0 = blockIdx.x * 16;            // 16 k-rows per workgroup
    const size_t base = ((size_t)b * BS) * LAYER + (size_t)b * BS;
    _Float16* dst = wsb + (size_t)b * (BS * BS);
    #pragma unroll
    for (int i = 0; i < 4; ++i) {
        const int idx = (i * 256 + threadIdx.x) * 4;   // over 16*256 elems
        const int k   = k0 + (idx >> 8);
        const int n   = idx & 255;
        const float4 v = *(const float4*)(blocks + base + (size_t)k * LAYER + n);
        _Float16* o = dst + k;                          // + n*BS below (transpose)
        o[(size_t)(n + 0) * BS] = (_Float16)tanhf(v.x);
        o[(size_t)(n + 1) * BS] = (_Float16)tanhf(v.y);
        o[(size_t)(n + 2) * BS] = (_Float16)tanhf(v.z);
        o[(size_t)(n + 3) * BS] = (_Float16)tanhf(v.w);
    }
}

// ---------------- kernel 2: y[:, b*256:(b+1)*256] = x[:, b*256:(b+1)*256] @ tanh(B_b)
__global__ __launch_bounds__(256, 1)
void block_diag_wmma_f16(const float* __restrict__ x,
                         const _Float16* __restrict__ wsb,
                         float* __restrict__ out)
{
    __shared__ _Float16 lds_a [M_TILE * LSTRIDE];   // 18432 B, x tile f16 [m][k]
    __shared__ _Float16 lds_bt[N_TILE * LSTRIDE];   // 36864 B, tanh(B)^T f16 [n][k]

    const int tid  = threadIdx.x;
    const int lane = tid & 31;
    const int w    = tid >> 5;        // wave 0..7
    const int r16  = lane & 15;
    const int hi   = lane >> 4;

    const int    row0 = blockIdx.x * M_TILE;
    const int    b    = blockIdx.y;
    const size_t bk0  = (size_t)b * BS;             // x column / output column base
    const _Float16* bsrc = wsb + (size_t)b * (BS * BS);

    v8f acc[16] = {};   // 16 rows x 256 cols per wave, f32 accumulate

    for (int kc = 0; kc < BS / K_CHUNK; ++kc) {
        const int k0 = kc * K_CHUNK;

#if USE_TDM
        // ---- stage B^T via Tensor Data Mover: one DMA of a 2D tile ----
        // global: 256 rows x 128B, row stride 512B  ->  LDS: row stride 144B
        // (pad 4 DWORDs after every 32 DWORDs), units of data_size = 8B.
        if (w == 0) {   // wave-uniform: only wave 0 issues the TDM op
            const unsigned long long ga = (unsigned long long)(size_t)(bsrc + k0);
            const unsigned ldsb = (unsigned)(size_t)(void*)&lds_bt[0];
            v4u g0;
            g0[0] = 1u;                                   // count=1, user descriptor
            g0[1] = ldsb;                                 // lds_addr (bytes)
            g0[2] = (unsigned)ga;                         // global_addr[31:0]
            g0[3] = (unsigned)(ga >> 32) | (2u << 30);    // global_addr[56:32] | type=2
            v8i g1;
            g1[0] = (3 << 16)      // data_size = 8 bytes
                  | (1 << 20)      // pad_enable
                  | (4 << 22)      // pad_interval: 32 DWORDs (=128B tile row)
                  | (3 << 25);     // pad_amount:   4 DWORDs (=16B -> 144B stride)
            g1[1] = 64  << 16;     // tensor_dim0 = 64 units (512B row)
            g1[2] = 256 << 16;     // tensor_dim1 = 256 rows
            g1[3] = 16  << 16;     // tile_dim0 = 16 units (128B)
            g1[4] = 256;           // tile_dim1 = 256 rows (tile_dim2 = 0)
            g1[5] = 64;            // tensor_dim0_stride = 64 units (512B)
            g1[6] = 0;             // tensor_dim1_stride = 0 (2D)
            g1[7] = 0;
            const v4i z4 = {0, 0, 0, 0};
            const v8i z8 = {0, 0, 0, 0, 0, 0, 0, 0};
            __builtin_amdgcn_tensor_load_to_lds(g0, g1, z4, z4, z8, 0);
        }
#else
        // ---- stage B^T: pure f16 copy from prepped ws: 256 rows x 64 halves ----
        #pragma unroll
        for (int i = 0; i < 16; ++i) {
            const int idx = i * 256 + tid;          // 4096 4-half chunks
            const int n   = idx >> 4;
            const int c   = (idx & 15) * 4;
            *(v4h*)(&lds_bt[n * LSTRIDE + c]) =
                *(const v4h*)(bsrc + (size_t)n * BS + k0 + c);
        }
#endif

        // ---- stage A: x[row0..+127][bk0+k0 ..+63] f32 -> f16 (row-major) ----
        #pragma unroll
        for (int i = 0; i < 8; ++i) {
            const int idx = (i * 256 + tid) * 4;    // over 128*64
            const int r   = idx >> 6;
            const int c   = idx & 63;
            const float4 v = *(const float4*)(x + (size_t)(row0 + r) * LAYER + bk0 + k0 + c);
            v4h h;
            h[0] = (_Float16)v.x; h[1] = (_Float16)v.y;
            h[2] = (_Float16)v.z; h[3] = (_Float16)v.w;
            *(v4h*)(&lds_a[r * LSTRIDE + c]) = h;   // 8B store, 8B-aligned
        }

#if USE_TDM
        __builtin_amdgcn_s_wait_tensorcnt(0);       // wave0's DMA done (others: cnt==0)
#endif
        __syncthreads();

        // ---- WMMA: wave w computes rows [w*16, w*16+16) x all 256 cols ----
        const _Float16* arow = &lds_a[(w * 16 + r16) * LSTRIDE];
        #pragma unroll
        for (int kk = 0; kk < K_CHUNK; kk += 32) {
            // A frag (16-bit A layout): halves 0-7 = K kk+8*hi+0..7, 8-15 = +16
            const v8h alo = *(const v8h*)(arow + kk + 8 * hi);
            const v8h ahi = *(const v8h*)(arow + kk + 16 + 8 * hi);
            v16h a;
            #pragma unroll
            for (int j = 0; j < 8; ++j) { a[j] = alo[j]; a[j + 8] = ahi[j]; }

            #pragma unroll
            for (int nt = 0; nt < 16; ++nt) {
                // B frag (16-bit B layout): lane=column, halves 0-7 = K kk+8*hi+0..7, 8-15 = +16
                const _Float16* brow =
                    &lds_bt[(nt * 16 + r16) * LSTRIDE + kk + 8 * hi];
                const v8h blo = *(const v8h*)(brow);
                const v8h bhi = *(const v8h*)(brow + 16);
                v16h bf;
                #pragma unroll
                for (int j = 0; j < 8; ++j) { bf[j] = blo[j]; bf[j + 8] = bhi[j]; }

                acc[nt] = __builtin_amdgcn_wmma_f32_16x16x32_f16(
                    /*neg_a=*/false, a, /*neg_b=*/false, bf,
                    /*c_mod=*/(short)0, acc[nt],
                    /*reuse_a=*/false, /*reuse_b=*/false);
            }
        }
        __syncthreads();
    }

    // ---- epilogue: C/D layout: VGPR v -> M = v + 8*hi, lane&15 -> N ----
    #pragma unroll
    for (int nt = 0; nt < 16; ++nt) {
        float* op = out + (size_t)(row0 + w * 16 + 8 * hi) * LAYER
                        + bk0 + nt * 16 + r16;
        #pragma unroll
        for (int v = 0; v < 8; ++v) {
            op[(size_t)v * LAYER] = acc[nt][v];
        }
    }
}

extern "C" void kernel_launch(void* const* d_in, const int* in_sizes, int n_in,
                              void* d_out, int out_size, void* d_ws, size_t ws_size,
                              hipStream_t stream) {
    const float* x      = (const float*)d_in[0];
    const float* blocks = (const float*)d_in[1];
    // d_in[2] = mask: unused — block-diagonal structure is static, tanh(0)=0
    float*     out = (float*)d_out;
    _Float16*  wsb = (_Float16*)d_ws;              // needs 16*256*256*2 = 2 MB

    // pass 1: tanh + transpose + f16 convert the 16 diagonal blocks
    prep_bt_f16<<<dim3(BS / 16, NBLK), 256, 0, stream>>>(blocks, wsb);

    // pass 2: 16 block GEMMs via WMMA (B staged by the Tensor Data Mover)
    const int n_rows = in_sizes[0] / LAYER;        // 8192
    dim3 grid(n_rows / M_TILE, NBLK);              // (64, 16)
    block_diag_wmma_f16<<<grid, 256, 0, stream>>>(x, wsb, out);
}
